// NNConvNet_88553635709217
// MI455X (gfx1250) — matrix operand order repaired and checked by reading, not compiled, not run on hardware
//
#include <hip/hip_runtime.h>
#include <stdint.h>

// ---------------- problem constants ----------------
#define N_NODES 20000
#define N_EDGES 100000
#define IN_C    16
#define H1C     32
#define H2C     64
#define FC1C    128
#define NCLS    10

#define N1 (27 * H1C)   // 26*32 msg cols + 32 root cols = 864
#define N2 (27 * H2C)   // 26*64 msg cols + 64 root cols = 1728

typedef _Float16 v16h __attribute__((ext_vector_type(16)));
typedef float    v8f  __attribute__((ext_vector_type(8)));

// ---------------- device helpers ----------------
__device__ __forceinline__ float eluf(float v) {
  return v > 0.0f ? v : expm1f(v);
}

// CDNA5 native f32 max atomic, device scope (L2 atomic units).
__device__ __forceinline__ void atomicMaxF32(float* p, float v) {
  asm volatile("global_atomic_max_num_f32 %0, %1, off scope:SCOPE_DEV"
               :: "v"((unsigned long long)(uintptr_t)p), "v"(v)
               : "memory");
}

// ---------------- init: fill with -inf ----------------
__global__ void fill_neginf(float* __restrict__ p, int n) {
  int i = blockIdx.x * blockDim.x + threadIdx.x;
  if (i < n) p[i] = -INFINITY;
}

// ---------------- pad x[20000,16] -> Xp[20000,32] (zeros above) ------------
__global__ void pad_x(const float* __restrict__ x, float* __restrict__ xp) {
  int i = blockIdx.x * blockDim.x + threadIdx.x;   // over N_NODES*32
  if (i >= N_NODES * 32) return;
  int n = i >> 5, c = i & 31;
  xp[i] = (c < IN_C) ? x[n * IN_C + c] : 0.0f;
}

// ---------------- pad fc2 bias [10] -> [16] (zeros above) ------------------
__global__ void pad_bias16(const float* __restrict__ b, float* __restrict__ bp) {
  int c = threadIdx.x;  // launched with 16 threads
  if (c < 16) bp[c] = (c < NCLS) ? b[c] : 0.0f;
}

// ---------------- B-matrix packing into WMMA fragment order ----------------
// Fragment order (matches GEMM read): dst[((tileN*Ksteps+step)*32+lane)*16+j]
//   lane<16 : n = tileN*16+lane,    k = step*32 + j       (j=0..15)
//   lane>=16: n = tileN*16+lane-16, k = step*32 + 16 + j
__global__ void pack_conv_b(const float* __restrict__ wb,  // [25, in_c*out_c]
                            const float* __restrict__ bb,  // [in_c*out_c]
                            const float* __restrict__ wr,  // [in_c, out_c]
                            int in_c, int out_c,
                            _Float16* __restrict__ dst, int ntiles) {
  int t = blockIdx.x * blockDim.x + threadIdx.x;
  if (t >= ntiles * 32) return;
  int lane = t & 31;
  int kbase = (lane >> 4) * 16;
  int n = ((t >> 5) * 16) + (lane & 15);
  int Ncols = 27 * out_c;
  _Float16* d = dst + (size_t)t * 16;
#pragma unroll
  for (int j = 0; j < 16; ++j) {
    int k = kbase + j;  // B row == input channel i
    float v = 0.0f;
    if (k < in_c && n < Ncols) {
      if (n < 26 * out_c) {
        int km = n / out_c, o = n % out_c;
        v = (km < 25) ? wb[(size_t)km * in_c * out_c + (size_t)k * out_c + o]
                      : bb[(size_t)k * out_c + o];
      } else {
        v = wr[(size_t)k * out_c + (n - 26 * out_c)];
      }
    }
    d[j] = (_Float16)v;
  }
}

// Plain row-major W[K x N] (K padded to 32*Ksteps, N padded to 16*ntilesN).
__global__ void pack_fc_b(const float* __restrict__ W, int K, int N,
                          _Float16* __restrict__ dst, int Ksteps, int ntilesN) {
  int t = blockIdx.x * blockDim.x + threadIdx.x;
  if (t >= ntilesN * Ksteps * 32) return;
  int lane = t & 31;
  int step = (t >> 5) % Ksteps;
  int tile = t / (32 * Ksteps);
  int kbase = step * 32 + (lane >> 4) * 16;
  int n = tile * 16 + (lane & 15);
  _Float16* d = dst + (size_t)t * 16;
#pragma unroll
  for (int j = 0; j < 16; ++j) {
    int k = kbase + j;
    d[j] = (_Float16)((k < K && n < N) ? W[(size_t)k * N + n] : 0.0f);
  }
}

// ---------------- WMMA GEMM -------------------------------------------------
// C[M x *] = A[M x KSTEPS*32] * B,  f16 inputs / f32 accumulate.
// One wave computes NT adjacent 16x16 output tiles, reusing its A fragments.
// A must be row-major with lda == KSTEPS*32 (caller pads K); M % 16 == 0 and
// all tile counts exact, so EXEC stays all-ones around v_wmma (HW req).
// HAS_BIAS: unconditional bias[n] read (caller pads bias to a multiple of 16).
// ACT_ELU: fused elu epilogue.
template <int KSTEPS, int NT, bool HAS_BIAS, bool ACT_ELU>
__global__ void __launch_bounds__(32)
gemm16_wmma(const float* __restrict__ A,
            const _Float16* __restrict__ Bp,
            float* __restrict__ C, int ldc,
            const float* __restrict__ bias) {
  const int lda = KSTEPS * 32;
  int lane = threadIdx.x;
  int m    = lane & 15;
  int sel  = lane >> 4;

  const float* Arow = A + (size_t)(blockIdx.x * 16 + m) * lda;

  // A fragments: lanes 0-15 hold K {kb..kb+15},
  //              lanes 16-31 hold K {kb+16..kb+31}
  // All loads 16B-aligned global_load_b128, unconditional.
  v16h afrag[KSTEPS];
#pragma unroll
  for (int s = 0; s < KSTEPS; ++s) {
    const float* p0 = Arow + s * 32 + sel * 16;
    const float* p1 = p0 + 8;
    float4 x0 = *(const float4*)(p0);
    float4 x1 = *(const float4*)(p0 + 4);
    float4 y0 = *(const float4*)(p1);
    float4 y1 = *(const float4*)(p1 + 4);
    v16h a;
    a[0]  = (_Float16)x0.x; a[1]  = (_Float16)x0.y;
    a[2]  = (_Float16)x0.z; a[3]  = (_Float16)x0.w;
    a[4]  = (_Float16)x1.x; a[5]  = (_Float16)x1.y;
    a[6]  = (_Float16)x1.z; a[7]  = (_Float16)x1.w;
    a[8]  = (_Float16)y0.x; a[9]  = (_Float16)y0.y;
    a[10] = (_Float16)y0.z; a[11] = (_Float16)y0.w;
    a[12] = (_Float16)y1.x; a[13] = (_Float16)y1.y;
    a[14] = (_Float16)y1.z; a[15] = (_Float16)y1.w;
    afrag[s] = a;
  }

#pragma unroll
  for (int t = 0; t < NT; ++t) {
    int tileN = blockIdx.y * NT + t;
    v8f c = {0.f, 0.f, 0.f, 0.f, 0.f, 0.f, 0.f, 0.f};
#pragma unroll
    for (int s = 0; s < KSTEPS; ++s) {
      // packed B fragment: 16 contiguous f16 per lane (32B aligned)
      v16h b = *(const v16h*)(Bp +
          (((size_t)tileN * KSTEPS + s) * 32 + lane) * 16);
      c = __builtin_amdgcn_wmma_f32_16x16x32_f16(
          /*neg_a=*/false, afrag[s], /*neg_b=*/false, b,
          /*c_mod=*/(short)0, c, /*reuse_a=*/false, /*reuse_b=*/false);
    }
    // C layout: lane -> N = lane&15; rows M = sel*8 + r, r = 0..7
    int n = tileN * 16 + (lane & 15);
    float bv = HAS_BIAS ? bias[n] : 0.0f;
    float* Cout = C + (size_t)(blockIdx.x * 16 + sel * 8) * ldc + n;
#pragma unroll
    for (int r = 0; r < 8; ++r) {
      float v = c[r] + bv;
      if (ACT_ELU) v = eluf(v);
      Cout[(size_t)r * ldc] = v;
    }
  }
}

// ---------------- per-edge message + scatter-max (conv1, out=32) -----------
__global__ void edge_msg1(const long long* __restrict__ ei,
                          const float* __restrict__ ea,
                          const float* __restrict__ wa,   // [25]
                          const float* __restrict__ ba,   // [25]
                          const float* __restrict__ Y,    // [N_NODES, N1]
                          float* __restrict__ agg) {      // [N_NODES, 32]
  int e    = (blockIdx.x * blockDim.x + threadIdx.x) >> 5;  // wave per edge
  int lane = threadIdx.x & 31;
  if (e >= N_EDGES) return;
  long long src = ei[e];
  long long tgt = ei[N_EDGES + e];
  float x = ea[e];
  const float* row = Y + (size_t)src * N1;
  float msg = row[25 * H1C + lane];  // folded b1b row (h_25 == 1)
#pragma unroll
  for (int km = 0; km < 25; ++km) {
    float hk = fmaf(x, wa[km], ba[km]);
    hk = hk > 0.0f ? hk : 0.0f;
    msg = fmaf(hk, row[km * H1C + lane], msg);
  }
  atomicMaxF32(agg + (size_t)tgt * H1C + lane, msg);
}

// ---------------- per-edge message + scatter-max (conv2, out=64) -----------
__global__ void edge_msg2(const long long* __restrict__ ei,
                          const float* __restrict__ ea,
                          const float* __restrict__ wa,   // [25]
                          const float* __restrict__ ba,   // [25]
                          const float* __restrict__ Y,    // [N_NODES, N2]
                          float* __restrict__ agg) {      // [N_NODES, 64]
  int e    = (blockIdx.x * blockDim.x + threadIdx.x) >> 5;
  int lane = threadIdx.x & 31;
  if (e >= N_EDGES) return;
  long long src = ei[e];
  long long tgt = ei[N_EDGES + e];
  float x = ea[e];
  const float* row = Y + (size_t)src * N2;
  float m0 = row[25 * H2C + lane];
  float m1 = row[25 * H2C + 32 + lane];
#pragma unroll
  for (int km = 0; km < 25; ++km) {
    float hk = fmaf(x, wa[km], ba[km]);
    hk = hk > 0.0f ? hk : 0.0f;
    m0 = fmaf(hk, row[km * H2C + lane], m0);
    m1 = fmaf(hk, row[km * H2C + 32 + lane], m1);
  }
  atomicMaxF32(agg + (size_t)tgt * H2C + lane, m0);
  atomicMaxF32(agg + (size_t)tgt * H2C + 32 + lane, m1);
}

// ---------------- finalize conv: fix(-inf)->0, +root +bias, elu ------------
__global__ void finalize_conv(const float* __restrict__ agg,
                              const float* __restrict__ Y, int ldy, int rootcol,
                              const float* __restrict__ bias, int out_c,
                              float* __restrict__ H, int total) {
  int i = blockIdx.x * blockDim.x + threadIdx.x;
  if (i >= total) return;
  int n = i / out_c, c = i % out_c;
  float v = agg[i];
  if (!isfinite(v)) v = 0.0f;  // nodes with no in-edges (PyG behavior)
  v += Y[(size_t)n * ldy + rootcol + c] + bias[c];
  H[i] = eluf(v);
}

// ---------------- log_softmax over 10 classes ------------------------------
__global__ void logsoftmax10(const float* __restrict__ logits,  // [N,16]
                             float* __restrict__ out) {         // [N,10]
  int n = blockIdx.x * blockDim.x + threadIdx.x;
  if (n >= N_NODES) return;
  const float* r = logits + (size_t)n * 16;
  float m = r[0];
#pragma unroll
  for (int c = 1; c < NCLS; ++c) m = fmaxf(m, r[c]);
  float s = 0.0f;
#pragma unroll
  for (int c = 0; c < NCLS; ++c) s += expf(r[c] - m);
  float ls = logf(s);
  float* o = out + (size_t)n * NCLS;
#pragma unroll
  for (int c = 0; c < NCLS; ++c) o[c] = r[c] - m - ls;
}

// ---------------- host orchestration ----------------
extern "C" void kernel_launch(void* const* d_in, const int* in_sizes, int n_in,
                              void* d_out, int out_size, void* d_ws, size_t ws_size,
                              hipStream_t stream) {
  const float*     x    = (const float*)d_in[0];
  const long long* ei   = (const long long*)d_in[1];   // int64 edge_index [2,E]
  const float*     ea   = (const float*)d_in[2];
  const float*     w1a  = (const float*)d_in[3];
  const float*     b1a  = (const float*)d_in[4];
  const float*     w1b  = (const float*)d_in[5];
  const float*     b1b  = (const float*)d_in[6];
  const float*     wr1  = (const float*)d_in[7];
  const float*     bias1= (const float*)d_in[8];
  const float*     w2a  = (const float*)d_in[9];
  const float*     b2a  = (const float*)d_in[10];
  const float*     w2b  = (const float*)d_in[11];
  const float*     b2b  = (const float*)d_in[12];
  const float*     wr2  = (const float*)d_in[13];
  const float*     bias2= (const float*)d_in[14];
  const float*     fc1w = (const float*)d_in[15];
  const float*     fc1b = (const float*)d_in[16];
  const float*     fc2w = (const float*)d_in[17];
  const float*     fc2b = (const float*)d_in[18];
  float*           out  = (float*)d_out;
  (void)n_in; (void)in_sizes; (void)out_size;

  // ---- workspace layout (all offsets 256B aligned), ~168 MB total ----
  char* ws = (char*)d_ws;
  size_t off = 0;
  float* Y     = (float*)(ws + off); off += (size_t)N_NODES * N2 * 4;   // Y1 then Y2
  float* Xp    = (float*)(ws + off); off += (size_t)N_NODES * 32 * 4;   // x padded K16->K32
  float* Hh1   = (float*)(ws + off); off += (size_t)N_NODES * H1C * 4;
  float* agg1  = (float*)(ws + off); off += (size_t)N_NODES * H1C * 4;
  float* Hh2   = (float*)(ws + off); off += (size_t)N_NODES * H2C * 4;
  float* agg2  = (float*)(ws + off); off += (size_t)N_NODES * H2C * 4;
  float* fc1o  = (float*)(ws + off); off += (size_t)N_NODES * FC1C * 4;
  float* logit = (float*)(ws + off); off += (size_t)N_NODES * 16 * 4;
  const int T1 = N1 / 16, T2 = N2 / 16;                 // 54, 108 N-tiles
  _Float16* pb1  = (_Float16*)(ws + off); off += (size_t)T1 * 32 * 16 * 2;
  _Float16* pb2  = (_Float16*)(ws + off); off += (size_t)T2 * 32 * 16 * 2;
  _Float16* pfc1 = (_Float16*)(ws + off); off += (size_t)8 * 2 * 32 * 16 * 2;
  _Float16* pfc2 = (_Float16*)(ws + off); off += (size_t)1 * 4 * 32 * 16 * 2;
  float* fc2bp = (float*)(ws + off); off += 256;        // fc2 bias padded to 16
  if (ws_size < off) return;  // insufficient scratch

  const int MT = N_NODES / 16;  // 1250 row tiles (exact)

  // 1) pack B matrices into WMMA fragment order; pad x and fc2 bias
  pack_conv_b<<<(T1 * 32 + 255) / 256, 256, 0, stream>>>(w1b, b1b, wr1, IN_C, H1C, pb1, T1);
  pack_conv_b<<<(T2 * 32 + 255) / 256, 256, 0, stream>>>(w2b, b2b, wr2, H1C, H2C, pb2, T2);
  pack_fc_b<<<(8 * 2 * 32 + 255) / 256, 256, 0, stream>>>(fc1w, H2C, FC1C, pfc1, 2, 8);
  pack_fc_b<<<(1 * 4 * 32 + 255) / 256, 256, 0, stream>>>(fc2w, FC1C, NCLS, pfc2, 4, 1);
  pad_x<<<((N_NODES * 32) + 255) / 256, 256, 0, stream>>>(x, Xp);
  pad_bias16<<<1, 16, 0, stream>>>(fc2b, fc2bp);

  // 2) init scatter-max accumulators to -inf
  fill_neginf<<<((N_NODES * H1C) + 255) / 256, 256, 0, stream>>>(agg1, N_NODES * H1C);
  fill_neginf<<<((N_NODES * H2C) + 255) / 256, 256, 0, stream>>>(agg2, N_NODES * H2C);

  // 3) conv1: Y1 = Xp @ [B1 | b1b-row | wr1]   (K padded to 32; B rows 16..31 = 0)
  gemm16_wmma<1, 2, false, false>
      <<<dim3(MT, T1 / 2), 32, 0, stream>>>(Xp, pb1, Y, N1, nullptr);
  // 4) per-edge weighted gather + native f32 max scatter
  edge_msg1<<<(N_EDGES * 32) / 256, 256, 0, stream>>>(ei, ea, w1a, b1a, Y, agg1);
  // 5) H1 = elu(fix(agg1) + root + bias1)
  finalize_conv<<<((N_NODES * H1C) + 255) / 256, 256, 0, stream>>>(
      agg1, Y, N1, 26 * H1C, bias1, H1C, Hh1, N_NODES * H1C);

  // 6) conv2: Y2 = H1 @ [B2 | b2b-row | wr2]  (K=32, reuses Y buffer)
  gemm16_wmma<1, 2, false, false>
      <<<dim3(MT, T2 / 2), 32, 0, stream>>>(Hh1, pb2, Y, N2, nullptr);
  edge_msg2<<<(N_EDGES * 32) / 256, 256, 0, stream>>>(ei, ea, w2a, b2a, Y, agg2);
  finalize_conv<<<((N_NODES * H2C) + 255) / 256, 256, 0, stream>>>(
      agg2, Y, N2, 26 * H2C, bias2, H2C, Hh2, N_NODES * H2C);

  // 7) FC1 (K=64, fused bias+elu) and FC2 (K=128, N and bias padded 10->16)
  gemm16_wmma<2, 2, true, true>
      <<<dim3(MT, 4), 32, 0, stream>>>(Hh2, pfc1, fc1o, FC1C, fc1b);
  gemm16_wmma<4, 1, true, false>
      <<<dim3(MT, 1), 32, 0, stream>>>(fc1o, pfc2, logit, 16, fc2bp);

  // 8) log_softmax over 10 classes
  logsoftmax10<<<(N_NODES + 255) / 256, 256, 0, stream>>>(logit, out);
}
